// SwinTransformerBlock_16733192585300
// MI455X (gfx1250) — matrix operand order repaired and checked by reading, not compiled, not run on hardware
//
#include <hip/hip_runtime.h>

// ---------------- problem constants ----------------
constexpr int BB = 16, HH = 56, WW = 56, CC = 384;
constexpr int HEADS = 12, WIN = 7, SHIFT = 3, HD = 32, HIDDEN = 1536;
constexpr int NWH = 8, NWW = 8, NW = 64, WA = 49;
constexpr int TOK = BB * NW * WA;          // 50176 rows (divisible by 128)
constexpr int QKVN = 3 * CC;               // 1152

typedef __bf16 v16bf __attribute__((ext_vector_type(16)));
typedef __bf16 v8bf  __attribute__((ext_vector_type(8)));
typedef float  v8f   __attribute__((ext_vector_type(8)));

__device__ __forceinline__ float gelu_f(float v) {
    return 0.5f * v * (1.0f + erff(v * 0.70710678118654752f));
}
__device__ __forceinline__ int zone_f(int g) { return g < 49 ? 0 : (g < 53 ? 1 : 2); }

// ---- DPP 16-lane xor-butterfly reductions (no LDS round trip) ----
// 0xB1 = quad_perm(1,0,3,2)  -> xor1
// 0x4E = quad_perm(2,3,0,1)  -> xor2
// 0x141 = ROW_HALF_MIRROR    -> combines quads within 8
// 0x140 = ROW_MIRROR         -> combines halves within 16-lane row
template<int CTRL>
__device__ __forceinline__ float dpp_f(float v) {
    int i = __builtin_bit_cast(int, v);
    return __builtin_bit_cast(float,
        __builtin_amdgcn_update_dpp(i, i, CTRL, 0xF, 0xF, true));
}
__device__ __forceinline__ float red16_max(float v) {
    v = fmaxf(v, dpp_f<0xB1>(v));
    v = fmaxf(v, dpp_f<0x4E>(v));
    v = fmaxf(v, dpp_f<0x141>(v));
    v = fmaxf(v, dpp_f<0x140>(v));
    return v;
}
__device__ __forceinline__ float red16_sum(float v) {
    v += dpp_f<0xB1>(v);
    v += dpp_f<0x4E>(v);
    v += dpp_f<0x141>(v);
    v += dpp_f<0x140>(v);
    return v;
}

// ---------------- weight transpose + bf16 convert: W[K,N] -> Wt[N,K] ----------------
__global__ void wtrans_kernel(const float* __restrict__ W, __bf16* __restrict__ Wt,
                              int K, int N) {
    size_t idx = (size_t)blockIdx.x * 256 + threadIdx.x;
    if (idx >= (size_t)K * N) return;
    int n = (int)(idx % N);
    int k = (int)(idx / N);
    Wt[(size_t)n * K + k] = (__bf16)W[idx];
}

// ---------------- precompute additive attention term: tab[nw][h][q][k] ----------------
__global__ void biastab_kernel(const float* __restrict__ rel_bias, float* __restrict__ tab) {
    size_t idx = (size_t)blockIdx.x * 256 + threadIdx.x;
    if (idx >= (size_t)NW * HEADS * WA * WA) return;
    int k  = (int)(idx % WA);
    int q  = (int)((idx / WA) % WA);
    int h  = (int)((idx / (WA * WA)) % HEADS);
    int nw = (int)(idx / ((size_t)WA * WA * HEADS));
    int wy = nw / NWW, wx = nw % NWW;
    int qy = q / WIN, qx = q % WIN, ky = k / WIN, kx = k % WIN;
    float add = rel_bias[((qy - ky + 6) * 13 + (qx - kx + 6)) * HEADS + h];
    int zq = zone_f(wy * WIN + qy) * 3 + zone_f(wx * WIN + qx);
    int zk = zone_f(wy * WIN + ky) * 3 + zone_f(wx * WIN + kx);
    if (zq != zk) add -= 100.0f;
    tab[idx] = add;
}

// ---------------- LayerNorm (+ optional shift/window remap) -> bf16 rows ----------------
__global__ __launch_bounds__(128) void ln_kernel(const float* __restrict__ src,
                                                 const float* __restrict__ g,
                                                 const float* __restrict__ bta,
                                                 __bf16* __restrict__ dst,
                                                 int shifted) {
    int tf = blockIdx.x;               // output token row
    size_t srow;
    if (shifted) {
        int bw = tf / WA, t = tf % WA;
        int b = bw / NW, nw = bw % NW;
        int wy = nw / NWW, wx = nw % NWW;
        int ty = t / WIN, tx = t % WIN;
        int sy = wy * WIN + ty, sx = wx * WIN + tx;
        int y = (sy + SHIFT) % HH, x = (sx + SHIFT) % WW;   // roll(x, -SHIFT)
        srow = ((size_t)(b * HH + y) * WW + x);
    } else {
        srow = (size_t)tf;
    }
    const float* s = src + srow * CC;
    __shared__ float red[4];
    int tid = threadIdx.x;
    float v0 = s[tid], v1 = s[tid + 128], v2 = s[tid + 256];
    float sum = v0 + v1 + v2;
    for (int m = 16; m >= 1; m >>= 1) sum += __shfl_xor(sum, m, 32);
    if ((tid & 31) == 0) red[tid >> 5] = sum;
    __syncthreads();
    float mean = (red[0] + red[1] + red[2] + red[3]) * (1.0f / 384.0f);
    float d0 = v0 - mean, d1 = v1 - mean, d2 = v2 - mean;
    float q = d0 * d0 + d1 * d1 + d2 * d2;
    for (int m = 16; m >= 1; m >>= 1) q += __shfl_xor(q, m, 32);
    __syncthreads();
    if ((tid & 31) == 0) red[tid >> 5] = q;
    __syncthreads();
    float var = (red[0] + red[1] + red[2] + red[3]) * (1.0f / 384.0f);
    float inv = rsqrtf(var + 1e-5f);
    __bf16* d = dst + (size_t)tf * CC;
    d[tid]       = (__bf16)(d0 * inv * g[tid]       + bta[tid]);
    d[tid + 128] = (__bf16)(d1 * inv * g[tid + 128] + bta[tid + 128]);
    d[tid + 256] = (__bf16)(d2 * inv * g[tid + 256] + bta[tid + 256]);
}

// ---------------- GEMM: A[M,K] bf16 x Bt[N,K] bf16 -> out[M,N] ----------------
// one wave computes a 16x64 strip (4 WMMAs sharing one A fragment per K-step);
// 8 waves/block -> 128x64 block tile
template<int ACT, int OUTF32, int RES>
__global__ __launch_bounds__(256) void gemm_bf16_kernel(const __bf16* __restrict__ A,
                                                        const __bf16* __restrict__ Bt,
                                                        const float* __restrict__ bias,
                                                        const float* __restrict__ resid,
                                                        void* __restrict__ outv,
                                                        int M, int N, int K) {
    int wave = threadIdx.x >> 5, lane = threadIdx.x & 31;
    int half = lane >> 4, l16 = lane & 15;
    int m0 = blockIdx.y * 128 + wave * 16;
    int n0 = blockIdx.x * 64;
    const __bf16* Ap = A + (size_t)(m0 + l16) * K;
    const __bf16* Bp[4];
#pragma unroll
    for (int nt = 0; nt < 4; ++nt)
        Bp[nt] = Bt + (size_t)(n0 + nt * 16 + l16) * K + half * 16;
    v8f acc[4] = {{}, {}, {}, {}};
    for (int k0 = 0; k0 < K; k0 += 32) {
        __builtin_prefetch(Ap + k0 + 128, 0, 1);
        v8bf alo = *reinterpret_cast<const v8bf*>(Ap + k0 + half * 16);
        v8bf ahi = *reinterpret_cast<const v8bf*>(Ap + k0 + half * 16 + 8);
        v16bf a = __builtin_shufflevector(alo, ahi, 0,1,2,3,4,5,6,7,8,9,10,11,12,13,14,15);
#pragma unroll
        for (int nt = 0; nt < 4; ++nt) {
            v16bf b = *reinterpret_cast<const v16bf*>(Bp[nt] + k0);
            acc[nt] = __builtin_amdgcn_wmma_f32_16x16x32_bf16(false, a, false, b, (short)0,
                                                              acc[nt], false, false);
        }
    }
#pragma unroll
    for (int nt = 0; nt < 4; ++nt) {
        float bs = bias[n0 + nt * 16 + l16];
#pragma unroll
        for (int r = 0; r < 8; ++r) {
            int m = m0 + r + half * 8;
            size_t base = (size_t)m * N + n0 + nt * 16 + l16;
            float v = acc[nt][r] + bs;
            if (ACT == 1) v = gelu_f(v);
            if (RES) v += resid[base];
            if (OUTF32) ((float*)outv)[base] = v;
            else        ((__bf16*)outv)[base] = (__bf16)v;
        }
    }
}

// ---------------- windowed attention: one block per (window, head) ----------------
__global__ __launch_bounds__(128) void attn_kernel(const __bf16* __restrict__ qkv,
                                                   const float* __restrict__ biastab,
                                                   __bf16* __restrict__ outp) {
    int bwh = blockIdx.x;
    int h = bwh % HEADS;
    int bw = bwh / HEADS;
    int nw = bw % NW;
    int tid = threadIdx.x;
    int wv = tid >> 5, lane = tid & 31, half = lane >> 4, l16 = lane & 15;

    __shared__ __attribute__((aligned(64))) float  tab_lds[WA][64];   // padded w/ -3e38
    __shared__ __attribute__((aligned(64))) __bf16 vT[HD][64];
    __shared__ __attribute__((aligned(64))) __bf16 p_lds[4][16][64];

    size_t base = (size_t)bw * WA * QKVN + h * HD;
    const float* tabsrc = biastab + ((size_t)nw * HEADS + h) * (WA * WA);
    for (int i = tid; i < WA * 64; i += 128) {
        int row = i >> 6, col = i & 63;
        tab_lds[row][col] = (col < WA) ? tabsrc[row * WA + col] : -3.0e38f;
    }
    for (int i = tid; i < 64 * HD; i += 128) {
        int tok = i >> 5, d = i & 31;
        int tc = tok > 48 ? 48 : tok;
        vT[d][tok] = qkv[base + 2 * CC + (size_t)tc * QKVN + d];
    }
    __syncthreads();

    // ---- S = q k^T (one 16x16x32 WMMA per 16x16 tile; wave owns 16 query rows) ----
    int m0 = wv * 16;
    int qm = m0 + l16;
    int qmc = qm > 48 ? 48 : qm;
    const __bf16* qrow = qkv + base + (size_t)qmc * QKVN;
    v8bf qlo = *reinterpret_cast<const v8bf*>(qrow + half * 16);
    v8bf qhi = *reinterpret_cast<const v8bf*>(qrow + half * 16 + 8);
    v16bf afrag = __builtin_shufflevector(qlo, qhi, 0,1,2,3,4,5,6,7,8,9,10,11,12,13,14,15);
    v8f s[4];
#pragma unroll
    for (int nt = 0; nt < 4; ++nt) {
        int kn = nt * 16 + l16;
        int knc = kn > 48 ? 48 : kn;
        const __bf16* krow = qkv + base + CC + (size_t)knc * QKVN + half * 16;
        v16bf bfrag = *reinterpret_cast<const v16bf*>(krow);
        v8f z = {};
        s[nt] = __builtin_amdgcn_wmma_f32_16x16x32_bf16(false, afrag, false, bfrag, (short)0,
                                                        z, false, false);
    }

    // ---- bias + shift-mask + softmax: branchless, DPP butterfly reductions ----
    const float scale = 0.17677669529663687f;   // 1/sqrt(32)
#pragma unroll
    for (int r = 0; r < 8; ++r) {
        int tq = m0 + r + 8 * half;
        int tqc = tq > 48 ? 48 : tq;
        const float* trow = &tab_lds[tqc][0];
        float mx = -3.0e38f;
#pragma unroll
        for (int nt = 0; nt < 4; ++nt) {
            float val = fmaf(s[nt][r], scale, trow[nt * 16 + l16]);
            s[nt][r] = val;
            mx = fmaxf(mx, val);
        }
        mx = red16_max(mx);
        float sum = 0.0f;
#pragma unroll
        for (int nt = 0; nt < 4; ++nt) {
            float e = __expf(s[nt][r] - mx);
            s[nt][r] = e;
            sum += e;
        }
        sum = red16_sum(sum);
        float inv = 1.0f / sum;
#pragma unroll
        for (int nt = 0; nt < 4; ++nt)
            p_lds[wv][r + 8 * half][nt * 16 + l16] = (__bf16)(s[nt][r] * inv);
    }
    __syncthreads();

    // ---- O = P v : two K-steps of 32 tokens, two 16-wide d tiles ----
    v8f o0 = {}, o1 = {};
#pragma unroll
    for (int ks = 0; ks < 2; ++ks) {
        const __bf16* prow = &p_lds[wv][l16][0];
        v8bf plo = *reinterpret_cast<const v8bf*>(prow + ks * 32 + half * 16);
        v8bf phi = *reinterpret_cast<const v8bf*>(prow + ks * 32 + half * 16 + 8);
        v16bf pa = __builtin_shufflevector(plo, phi, 0,1,2,3,4,5,6,7,8,9,10,11,12,13,14,15);
        v16bf vb0 = *reinterpret_cast<const v16bf*>(&vT[l16][ks * 32 + half * 16]);
        v16bf vb1 = *reinterpret_cast<const v16bf*>(&vT[16 + l16][ks * 32 + half * 16]);
        o0 = __builtin_amdgcn_wmma_f32_16x16x32_bf16(false, pa, false, vb0, (short)0, o0, false, false);
        o1 = __builtin_amdgcn_wmma_f32_16x16x32_bf16(false, pa, false, vb1, (short)0, o1, false, false);
    }
#pragma unroll
    for (int r = 0; r < 8; ++r) {
        int m = m0 + r + 8 * half;
        if (m < WA) {
            size_t orow = ((size_t)bw * WA + m) * CC + h * HD;
            outp[orow + l16]      = (__bf16)o0[r];
            outp[orow + 16 + l16] = (__bf16)o1[r];
        }
    }
}

// ---------------- window reverse + unshift + residual: x2 = x + roll(win_rev(proj)) ----------------
__global__ void resid_kernel(const float* __restrict__ x, const float* __restrict__ projout,
                             float* __restrict__ x2) {
    size_t idx = (size_t)blockIdx.x * 256 + threadIdx.x;
    if (idx >= (size_t)TOK * CC) return;
    int c = (int)(idx % CC);
    int row = (int)(idx / CC);
    int xcol = row % WW;
    int y = (row / WW) % HH;
    int b = row / (HH * WW);
    int sy = (y + HH - SHIFT) % HH, sx = (xcol + WW - SHIFT) % WW;   // roll(+SHIFT)
    int wy = sy / WIN, ty = sy % WIN, wx = sx / WIN, tx = sx % WIN;
    int prow = (b * NW + wy * NWW + wx) * WA + ty * WIN + tx;
    x2[idx] = x[idx] + projout[(size_t)prow * CC + c];
}

// ---------------- host launch ----------------
extern "C" void kernel_launch(void* const* d_in, const int* in_sizes, int n_in,
                              void* d_out, int out_size, void* d_ws, size_t ws_size,
                              hipStream_t stream) {
    const float* x      = (const float*)d_in[0];
    const float* n1g    = (const float*)d_in[1];
    const float* n1b    = (const float*)d_in[2];
    const float* qkv_w  = (const float*)d_in[3];
    const float* qkv_b  = (const float*)d_in[4];
    const float* rel_b  = (const float*)d_in[5];
    const float* proj_w = (const float*)d_in[6];
    const float* proj_b = (const float*)d_in[7];
    const float* n2g    = (const float*)d_in[8];
    const float* n2b    = (const float*)d_in[9];
    const float* fc1_w  = (const float*)d_in[10];
    const float* fc1_b  = (const float*)d_in[11];
    const float* fc2_w  = (const float*)d_in[12];
    const float* fc2_b  = (const float*)d_in[13];
    float* out = (float*)d_out;

    char* ws = (char*)d_ws;
    auto au = [](size_t v) { return (v + 255) & ~(size_t)255; };
    size_t o = 0;
    __bf16* qkv_wT  = (__bf16*)(ws + o); o += au((size_t)QKVN * CC * 2);
    __bf16* proj_wT = (__bf16*)(ws + o); o += au((size_t)CC * CC * 2);
    __bf16* fc1_wT  = (__bf16*)(ws + o); o += au((size_t)HIDDEN * CC * 2);
    __bf16* fc2_wT  = (__bf16*)(ws + o); o += au((size_t)CC * HIDDEN * 2);
    float*  btab    = (float*)(ws + o);  o += au((size_t)NW * HEADS * WA * WA * 4);
    __bf16* regA    = (__bf16*)(ws + o); o += au((size_t)TOK * CC * 2);      // xw, later h_in
    char*   regB    = ws + o;            o += au((size_t)TOK * QKVN * 2);    // qkv, later projout
    __bf16* regC    = (__bf16*)(ws + o); o += au((size_t)TOK * CC * 2);      // attn out
    float*  x2      = (float*)(ws + o);  o += au((size_t)TOK * CC * 4);
    __bf16* h1      = (__bf16*)(ws + o); o += au((size_t)TOK * HIDDEN * 2);
    (void)ws_size; (void)in_sizes; (void)n_in; (void)out_size;

    __bf16* qkvbuf  = (__bf16*)regB;
    float*  projout = (float*)regB;

    // weights -> transposed bf16, bias table
    {
        size_t n;
        n = (size_t)CC * QKVN;
        wtrans_kernel<<<(unsigned)((n + 255) / 256), 256, 0, stream>>>(qkv_w, qkv_wT, CC, QKVN);
        n = (size_t)CC * CC;
        wtrans_kernel<<<(unsigned)((n + 255) / 256), 256, 0, stream>>>(proj_w, proj_wT, CC, CC);
        n = (size_t)CC * HIDDEN;
        wtrans_kernel<<<(unsigned)((n + 255) / 256), 256, 0, stream>>>(fc1_w, fc1_wT, CC, HIDDEN);
        n = (size_t)HIDDEN * CC;
        wtrans_kernel<<<(unsigned)((n + 255) / 256), 256, 0, stream>>>(fc2_w, fc2_wT, HIDDEN, CC);
        n = (size_t)NW * HEADS * WA * WA;
        biastab_kernel<<<(unsigned)((n + 255) / 256), 256, 0, stream>>>(rel_b, btab);
    }

    // LN1 + roll + window partition -> bf16
    ln_kernel<<<TOK, 128, 0, stream>>>(x, n1g, n1b, regA, 1);

    // QKV GEMM  [TOK,384] x [1152,384]^T
    gemm_bf16_kernel<0, 0, 0><<<dim3(QKVN / 64, TOK / 128), 256, 0, stream>>>(
        regA, qkv_wT, qkv_b, nullptr, (void*)qkvbuf, TOK, QKVN, CC);

    // attention
    attn_kernel<<<BB * NW * HEADS, 128, 0, stream>>>(qkvbuf, btab, regC);

    // proj GEMM -> f32
    gemm_bf16_kernel<0, 1, 0><<<dim3(CC / 64, TOK / 128), 256, 0, stream>>>(
        regC, proj_wT, proj_b, nullptr, (void*)projout, TOK, CC, CC);

    // window reverse + unshift + residual
    {
        size_t n = (size_t)TOK * CC;
        resid_kernel<<<(unsigned)((n + 255) / 256), 256, 0, stream>>>(x, projout, x2);
    }

    // LN2 -> bf16
    ln_kernel<<<TOK, 128, 0, stream>>>(x2, n2g, n2b, regA, 0);

    // FC1 GEMM + GELU -> bf16
    gemm_bf16_kernel<1, 0, 0><<<dim3(HIDDEN / 64, TOK / 128), 256, 0, stream>>>(
        regA, fc1_wT, fc1_b, nullptr, (void*)h1, TOK, HIDDEN, CC);

    // FC2 GEMM + bias + residual -> d_out (f32)
    gemm_bf16_kernel<0, 1, 1><<<dim3(CC / 64, TOK / 128), 256, 0, stream>>>(
        h1, fc2_wT, fc2_b, x2, (void*)out, TOK, CC, HIDDEN);
}
